// GSM_23124103921699
// MI455X (gfx1250) — compile-verified, matrix operand order checked by reading, not run on hardware
//
#include <hip/hip_runtime.h>
#include <hip/hip_bf16.h>
#include <math.h>

#define B_ 4
#define S_ 4096
#define D_ 1024
#define G_ 8
#define R_ 32
#define DG_ 128
#define FMAX_ 2049

typedef _Float16 v8h  __attribute__((ext_vector_type(8)));
typedef _Float16 v16h __attribute__((ext_vector_type(16)));
typedef float    v8f  __attribute__((ext_vector_type(8)));
typedef unsigned int u32x4 __attribute__((ext_vector_type(4)));
typedef int          i32x8 __attribute__((ext_vector_type(8)));
typedef int          i32x4 __attribute__((ext_vector_type(4)));

// ---------------------------------------------------------------------------
// WMMA helper (CDNA5 wave32). 16x16x32 f16 -> f32 accumulate.
// ---------------------------------------------------------------------------
__device__ __forceinline__ v8f wmma_f32_f16(v16h a, v16h b, v8f c) {
    return __builtin_amdgcn_wmma_f32_16x16x32_f16(
        /*neg_a=*/false, a, /*neg_b=*/false, b,
        /*c_mod=*/(short)0, c, /*reuse_a=*/false, /*reuse_b=*/false);
}

// Load a 16x32 fragment from LDS (two 16B ds_load_b128 per fragment).
// A operand: lds[M*ld + K] row-major. B operand: pass an LDS matrix where
// lds[N*ld + K] holds element (K,N) (col-major data or symmetric DFT matrix).
__device__ __forceinline__ v16h ld_frag(const _Float16* lds, int m0, int k0, int ld) {
    int lane = threadIdx.x & 31;
    int M  = m0 + (lane & 15);
    int kb = k0 + ((lane & 16) ? 8 : 0);
    const v8h* p0 = (const v8h*)(lds + M * ld + kb);
    const v8h* p1 = (const v8h*)(lds + M * ld + kb + 16);
    v8h lo = *p0;
    v8h hi = *p1;
    v16h r;
#pragma unroll
    for (int i = 0; i < 8; ++i) { r[i] = lo[i]; r[i + 8] = hi[i]; }
    return r;
}

// ---------------------------------------------------------------------------
// Tensor Data Mover: contiguous 8 KB (2048 dword) tile between global and LDS.
// Descriptor per CDNA5 ISA (D# group0/group1 bit layout).
// This toolchain's builtin takes 6 args: (g0, g1, g2, g3, g4, cpol).
// ---------------------------------------------------------------------------
__device__ __forceinline__ u32x4 tdm_g0(unsigned lds_addr, const void* gp) {
    unsigned long long ga = (unsigned long long)gp;
    u32x4 g0;
    g0.x = 1u;                                            // count=1, user mode
    g0.y = lds_addr;                                      // lds_addr [63:32]
    g0.z = (unsigned)(ga & 0xFFFFFFFFu);                  // global_addr [95:64]
    g0.w = (unsigned)((ga >> 32) & 0x1FFFFFFu) | (2u << 30); // addr hi + type=2
    return g0;
}
__device__ __forceinline__ i32x8 tdm_g1_8k() {
    i32x8 g1;
    g1[0] = (int)(2u << 16);      // workgroup_mask=0, data_size=2 (4 bytes)
    g1[1] = (int)(2048u << 16);   // tensor_dim0 = 2048 dwords (bits 63:48)
    g1[2] = (int)(1u << 16);      // tensor_dim0 hi=0, tensor_dim1 = 1
    g1[3] = (int)(2048u << 16);   // tensor_dim1 hi=0, tile_dim0 = 2048
    g1[4] = 1;                    // tile_dim1 = 1, tile_dim2 = 0
    g1[5] = 2048;                 // tensor_dim0_stride = 2048
    g1[6] = 0;
    g1[7] = 0;
    return g1;
}
__device__ __forceinline__ void tdm_load_8k(unsigned lds_addr, const void* gp) {
    i32x4 z4 = {0, 0, 0, 0};
    i32x8 z8 = {0, 0, 0, 0, 0, 0, 0, 0};
    __builtin_amdgcn_tensor_load_to_lds(tdm_g0(lds_addr, gp), tdm_g1_8k(),
                                        z4, z4, z8, 0);
}
__device__ __forceinline__ void tdm_store_8k(unsigned lds_addr, void* gp) {
    i32x4 z4 = {0, 0, 0, 0};
    i32x8 z8 = {0, 0, 0, 0, 0, 0, 0, 0};
    __builtin_amdgcn_tensor_store_from_lds(tdm_g0(lds_addr, gp), tdm_g1_8k(),
                                           z4, z4, z8, 0);
}

// ---------------------------------------------------------------------------
// 0) zero the sbar accumulator
// ---------------------------------------------------------------------------
__global__ void gsm_zero(float* p, int n) {
    int i = blockIdx.x * blockDim.x + threadIdx.x;
    if (i < n) p[i] = 0.f;
}

// ---------------------------------------------------------------------------
// 1) per-(b,s) RMS: rms[r] = rsqrt(mean(x^2)+eps)
// ---------------------------------------------------------------------------
__global__ void __launch_bounds__(256) gsm_rms(const float* __restrict__ x,
                                               float* __restrict__ rms) {
    __shared__ float red[256];
    int r = blockIdx.x;
    int t = threadIdx.x;
    const float* row = x + (size_t)r * D_;
    float v = 0.f;
#pragma unroll
    for (int j = 0; j < 4; ++j) { float xv = row[t + 256 * j]; v += xv * xv; }
    red[t] = v;
    __syncthreads();
    for (int s2 = 128; s2 > 0; s2 >>= 1) {
        if (t < s2) red[t] += red[t + s2];
        __syncthreads();
    }
    if (t == 0) rms[r] = rsqrtf(red[0] * (1.f / D_) + 1e-6f);
}

// ---------------------------------------------------------------------------
// 2) normalize, transpose to x_t[b][d][s] (f16), fused mean-pool accumulation
// ---------------------------------------------------------------------------
__global__ void __launch_bounds__(256) gsm_norm_t(const float* __restrict__ x,
                                                  const float* __restrict__ nw,
                                                  const float* __restrict__ rms,
                                                  _Float16* __restrict__ xt,
                                                  float* __restrict__ sbar) {
    int d0 = blockIdx.x * 64;
    int s0 = blockIdx.y * 64;
    int b  = blockIdx.z;
    __shared__ _Float16 tile[64][65];
    __shared__ float red[256];
    int t = threadIdx.x;
    int j = t & 63;
    int i0 = t >> 6;
    float w = nw[d0 + j];
    float psum = 0.f;
#pragma unroll
    for (int ii = 0; ii < 16; ++ii) {
        int i = i0 + 4 * ii;
        int s = s0 + i;
        float v = x[((size_t)(b * S_ + s)) * D_ + d0 + j] * rms[b * S_ + s] * w;
        tile[i][j] = (_Float16)v;
        psum += v;
    }
    red[t] = psum;
    __syncthreads();
    if (t < 64) {
        float ssum = red[t] + red[t + 64] + red[t + 128] + red[t + 192];
        atomicAdd(&sbar[b * D_ + d0 + t], ssum * (1.f / S_));
    }
    int sl = t & 63;
    int dr0 = t >> 6;
#pragma unroll
    for (int ii = 0; ii < 16; ++ii) {
        int dr = dr0 + 4 * ii;
        xt[((size_t)(b * D_ + d0 + dr)) * S_ + s0 + sl] = tile[sl][dr];
    }
}

// ---------------------------------------------------------------------------
// 3) gate = sigmoid(sbar @ pool_W + b);  scale[bg][f] = 1 + 0.5 * (gate . U)
// ---------------------------------------------------------------------------
__global__ void __launch_bounds__(256) gsm_gate_scale(const float* __restrict__ sbar,
                                                      const float* __restrict__ poolW,
                                                      const float* __restrict__ poolb,
                                                      const float* __restrict__ U,
                                                      float* __restrict__ scaleBuf) {
    int bg = blockIdx.x;
    int b = bg >> 3, g = bg & 7;
    int t = threadIdx.x;
    __shared__ float red[256];
    __shared__ float gateL[R_];
    int r = t >> 3, part = t & 7;
    const float* sb = sbar + b * D_;
    float acc = 0.f;
    for (int d = part * 128; d < part * 128 + 128; ++d)
        acc += sb[d] * poolW[(size_t)d * (G_ * R_) + g * R_ + r];
    red[t] = acc;
    __syncthreads();
    if (part == 0) {
        float s = 0.f;
#pragma unroll
        for (int q = 0; q < 8; ++q) s += red[r * 8 + q];
        s += poolb[g * R_ + r];
        gateL[r] = 1.f / (1.f + __expf(-s));
    }
    __syncthreads();
    for (int f = t; f < FMAX_; f += 256) {
        float m = 0.f;
#pragma unroll 8
        for (int rr = 0; rr < R_; ++rr)
            m += gateL[rr] * U[((size_t)(g * R_ + rr)) * FMAX_ + f];
        scaleBuf[bg * FMAX_ + f] = 1.f + 0.5f * m;
    }
}

// ---------------------------------------------------------------------------
// 4) Spectral filter via tensor-core FFT (4096 = 64x64 Cooley-Tukey).
//    TDM async tile in/out; four complex 64x64x64 WMMA matmul stages with
//    incremental twiddle rotation on the f32 accumulators.
// ---------------------------------------------------------------------------
__global__ void __launch_bounds__(256) gsm_spectral(const _Float16* __restrict__ xt,
                                                    _Float16* __restrict__ yt,
                                                    const float* __restrict__ Wr,
                                                    const float* __restrict__ Wi,
                                                    const float* __restrict__ scaleBuf) {
    int c  = blockIdx.x;
    int bg = blockIdx.y;
    int b = bg >> 3, g = bg & 7;
    int d = g * DG_ + c;
    const size_t base = ((size_t)(b * D_ + d)) * S_;

    __shared__ _Float16 sCc[4096], sCs[4096], sCn[4096];  // cos, sin, -sin
    __shared__ _Float16 Qr[4096], Qi[4096], Pr[4096], Pi[4096];

    int t = threadIdx.x;
    int lane = t & 31;
    int wave = t >> 5;

    // TDM: async DMA the contiguous 8 KB input tile into Qi (linear layout).
    if (wave == 0) {
        tdm_load_8k((unsigned)(size_t)(void*)&Qi[0], (const void*)(xt + base));
    }

    // Build DFT-64 matrices while the DMA is in flight.
#pragma unroll
    for (int ii = 0; ii < 16; ++ii) {
        int idx = t + 256 * ii;
        int n = idx >> 6, k = idx & 63;
        float sv, cv;
        __sincosf(0.09817477042f * (float)(n * k), &sv, &cv);  // 2*pi/64
        sCc[idx] = (_Float16)cv;
        sCs[idx] = (_Float16)sv;
        sCn[idx] = (_Float16)(-sv);
    }
    if (wave == 0) __builtin_amdgcn_s_wait_tensorcnt(0);
    __syncthreads();

    // Transpose staging -> Pr col-major (B operand of stage 1).
#pragma unroll
    for (int ii = 0; ii < 16; ++ii) {
        int idx = t + 256 * ii;
        Pr[(idx & 63) * 64 + (idx >> 6)] = Qi[idx];
    }
    __syncthreads();

    const float TWN = 6.28318530718f / 4096.f;
    int colN = lane & 15;
    int rowB = (lane & 16) ? 8 : 0;

    // ---- Stage 1: X1 = D64 * T (real input), then twiddle W4096^{k1*n2}
    for (int tt = 0; tt < 2; ++tt) {
        int tid = wave * 2 + tt, tm = tid >> 2, tn = tid & 3;
        v8f ar = {}, ai = {};
#pragma unroll
        for (int kc = 0; kc < 64; kc += 32) {
            v16h aC = ld_frag(sCc, tm * 16, kc, 64);
            v16h aN = ld_frag(sCn, tm * 16, kc, 64);
            v16h bT = ld_frag(Pr,  tn * 16, kc, 64);
            ar = wmma_f32_f16(aC, bT, ar);
            ai = wmma_f32_f16(aN, bT, ai);
        }
        int N = tn * 16 + colN;
        int M0 = tm * 16 + rowB;
        float cv, sv, cd, sd;
        __sincosf(TWN * (float)(M0 * N), &sv, &cv);
        __sincosf(TWN * (float)N, &sd, &cd);          // per-row rotation step
#pragma unroll
        for (int v = 0; v < 8; ++v) {
            int M = M0 + v;
            float br = ar[v] * cv + ai[v] * sv;       // (a+bi)(cv - i sv)
            float bi = ai[v] * cv - ar[v] * sv;
            Qr[M * 64 + N] = (_Float16)br;
            Qi[M * 64 + N] = (_Float16)bi;
            float cn = cv * cd - sv * sd;             // rotate twiddle
            sv = sv * cd + cv * sd;
            cv = cn;
        }
    }
    __syncthreads();

    // ---- Stage 2: X2 = X1 * D64; apply Hermitian-extended filter * scale
    for (int tt = 0; tt < 2; ++tt) {
        int tid = wave * 2 + tt, tm = tid >> 2, tn = tid & 3;
        v8f xr = {}, xi = {};
#pragma unroll
        for (int kc = 0; kc < 64; kc += 32) {
            v16h aR = ld_frag(Qr, tm * 16, kc, 64);
            v16h aI = ld_frag(Qi, tm * 16, kc, 64);
            v16h bC = ld_frag(sCc, tn * 16, kc, 64);
            v16h bS = ld_frag(sCs, tn * 16, kc, 64);
            v16h bN = ld_frag(sCn, tn * 16, kc, 64);
            xr = wmma_f32_f16(aR, bC, xr);
            xr = wmma_f32_f16(aI, bS, xr);
            xi = wmma_f32_f16(aI, bC, xi);
            xi = wmma_f32_f16(aR, bN, xi);
        }
        int N = tn * 16 + colN;
#pragma unroll
        for (int v = 0; v < 8; ++v) {
            int M = tm * 16 + rowB + v;
            int k = M + 64 * N;                       // frequency index
            int f = (k <= 2048) ? k : 4096 - k;
            float sg = (k <= 2048) ? 1.f : -1.f;      // conjugate mirror
            float sc = scaleBuf[bg * FMAX_ + f];
            float hr = Wr[f * DG_ + c] * sc;
            float hi = sg * Wi[f * DG_ + c] * sc;
            float yr2 = xr[v] * hr - xi[v] * hi;
            float yi2 = xr[v] * hi + xi[v] * hr;
            Pr[M * 64 + N] = (_Float16)yr2;
            Pi[M * 64 + N] = (_Float16)yi2;
        }
    }
    __syncthreads();

    // ---- Stage 3: Z = Y * conj(D64); conj twiddle; store col-major
    for (int tt = 0; tt < 2; ++tt) {
        int tid = wave * 2 + tt, tm = tid >> 2, tn = tid & 3;
        v8f zr = {}, zi = {};
#pragma unroll
        for (int kc = 0; kc < 64; kc += 32) {
            v16h aR = ld_frag(Pr, tm * 16, kc, 64);
            v16h aI = ld_frag(Pi, tm * 16, kc, 64);
            v16h bC = ld_frag(sCc, tn * 16, kc, 64);
            v16h bS = ld_frag(sCs, tn * 16, kc, 64);
            v16h bN = ld_frag(sCn, tn * 16, kc, 64);
            zr = wmma_f32_f16(aR, bC, zr);
            zr = wmma_f32_f16(aI, bN, zr);
            zi = wmma_f32_f16(aI, bC, zi);
            zi = wmma_f32_f16(aR, bS, zi);
        }
        int N = tn * 16 + colN;
        int M0 = tm * 16 + rowB;
        float cv, sv, cd, sd;
        __sincosf(TWN * (float)(M0 * N), &sv, &cv);
        __sincosf(TWN * (float)N, &sd, &cd);
#pragma unroll
        for (int v = 0; v < 8; ++v) {
            int M = M0 + v;
            float br = zr[v] * cv - zi[v] * sv;       // (a+bi)(cv + i sv)
            float bi = zi[v] * cv + zr[v] * sv;
            Qr[N * 64 + M] = (_Float16)br;            // column-major
            Qi[N * 64 + M] = (_Float16)bi;
            float cn = cv * cd - sv * sd;
            sv = sv * cd + cv * sd;
            cv = cn;
        }
    }
    __syncthreads();

    // ---- Stage 4: y = (1/4096) Re( conj(D64) * Z' ), to LDS linear (s order)
    for (int tt = 0; tt < 2; ++tt) {
        int tid = wave * 2 + tt, tm = tid >> 2, tn = tid & 3;
        v8f yr = {};
#pragma unroll
        for (int kc = 0; kc < 64; kc += 32) {
            v16h aC = ld_frag(sCc, tm * 16, kc, 64);
            v16h aN = ld_frag(sCn, tm * 16, kc, 64);
            v16h bR = ld_frag(Qr, tn * 16, kc, 64);
            v16h bI = ld_frag(Qi, tn * 16, kc, 64);
            yr = wmma_f32_f16(aC, bR, yr);
            yr = wmma_f32_f16(aN, bI, yr);
        }
        int N = tn * 16 + colN;
#pragma unroll
        for (int v = 0; v < 8; ++v) {
            int M = tm * 16 + rowB + v;               // s = 64*M + N
            Pr[M * 64 + N] = (_Float16)(yr[v] * (1.f / 4096.f));
        }
    }
    __syncthreads();

    // TDM: async DMA the 8 KB result tile from LDS back to y_t.
    if (wave == 0) {
        tdm_store_8k((unsigned)(size_t)(void*)&Pr[0], (void*)(yt + base));
        __builtin_amdgcn_s_wait_tensorcnt(0);         // keep WG alive until done
    }
}

// ---------------------------------------------------------------------------
// 5) Projection GEMM: out[16384,1024] = y[16384,1024] @ proj_W[1024,1024]
// ---------------------------------------------------------------------------
__global__ void __launch_bounds__(256) gsm_proj(const _Float16* __restrict__ yt,
                                                const float* __restrict__ pw,
                                                float* __restrict__ out) {
    int n0   = blockIdx.x * 64;
    int row0 = blockIdx.y * 64;
    int b  = row0 >> 12;
    int s0 = row0 & 4095;
    __shared__ _Float16 As[64 * 72];   // [m][k], ld=72 keeps 16B alignment
    __shared__ _Float16 Bs[64 * 72];   // [n][k]
    int t = threadIdx.x, lane = t & 31, wave = t >> 5;
    v8f acc[2];
    acc[0] = (v8f){};
    acc[1] = (v8f){};
    for (int kk = 0; kk < D_; kk += 64) {
#pragma unroll
        for (int ii = 0; ii < 16; ++ii) {
            int idx = t + 256 * ii;
            int ml = idx & 63, kl = idx >> 6;
            As[ml * 72 + kl] = yt[((size_t)(b * D_ + kk + kl)) * S_ + s0 + ml];
            Bs[ml * 72 + kl] = (_Float16)pw[((size_t)(kk + kl)) * D_ + n0 + ml];
        }
        if (kk + 64 < D_)
            __builtin_prefetch(&pw[((size_t)(kk + 64)) * D_ + n0], 0, 1);
        __syncthreads();
#pragma unroll
        for (int tt = 0; tt < 2; ++tt) {
            int tid = wave * 2 + tt, tm = tid >> 2, tn = tid & 3;
#pragma unroll
            for (int kc = 0; kc < 64; kc += 32) {
                v16h a  = ld_frag(As, tm * 16, kc, 72);
                v16h bb = ld_frag(Bs, tn * 16, kc, 72);
                acc[tt] = wmma_f32_f16(a, bb, acc[tt]);
            }
        }
        __syncthreads();
    }
    int colN = lane & 15;
    int rowB = (lane & 16) ? 8 : 0;
#pragma unroll
    for (int tt = 0; tt < 2; ++tt) {
        int tid = wave * 2 + tt, tm = tid >> 2, tn = tid & 3;
#pragma unroll
        for (int v = 0; v < 8; ++v) {
            int M = tm * 16 + rowB + v;
            int N = tn * 16 + colN;
            out[((size_t)(row0 + M)) * D_ + n0 + N] = acc[tt][v];
        }
    }
}

// ---------------------------------------------------------------------------
// launcher
// ---------------------------------------------------------------------------
extern "C" void kernel_launch(void* const* d_in, const int* in_sizes, int n_in,
                              void* d_out, int out_size, void* d_ws, size_t ws_size,
                              hipStream_t stream) {
    const float* x      = (const float*)d_in[0];
    const float* norm_w = (const float*)d_in[1];
    const float* Wr     = (const float*)d_in[2];
    const float* Wi     = (const float*)d_in[3];
    const float* poolW  = (const float*)d_in[4];
    const float* poolb  = (const float*)d_in[5];
    const float* U      = (const float*)d_in[6];
    const float* projW  = (const float*)d_in[7];
    float* out = (float*)d_out;

    char* ws = (char*)d_ws;
    _Float16* xt      = (_Float16*)(ws + 0);                    // 32 MB
    _Float16* yt      = (_Float16*)(ws + (size_t)33554432);     // 32 MB
    float*    rms     = (float*)(ws + (size_t)67108864);        // 64 KB
    float*    sbar    = (float*)(ws + (size_t)67174400);        // 16 KB
    float*    scaleB  = (float*)(ws + (size_t)67190784);        // ~256 KB

    gsm_zero<<<dim3(16), 256, 0, stream>>>(sbar, B_ * D_);
    gsm_rms<<<dim3(B_ * S_), 256, 0, stream>>>(x, rms);
    gsm_norm_t<<<dim3(D_ / 64, S_ / 64, B_), 256, 0, stream>>>(x, norm_w, rms, xt, sbar);
    gsm_gate_scale<<<dim3(B_ * G_), 256, 0, stream>>>(sbar, poolW, poolb, U, scaleB);
    gsm_spectral<<<dim3(DG_, B_ * G_), 256, 0, stream>>>(xt, yt, Wr, Wi, scaleB);
    gsm_proj<<<dim3(D_ / 64, (B_ * S_) / 64), 256, 0, stream>>>(yt, projW, out);
}
